// CTRNN_34299608825962
// MI455X (gfx1250) — matrix operand order repaired and checked by reading, not compiled, not run on hardware
//
#include <hip/hip_runtime.h>

// CTRNN on gfx1250: bf16 WMMA with f32 accumulation.
// T=512, B=128, I=256, H=512, alpha = 0.2.
//
// Weights are pre-shuffled into WMMA-fragment-major layout so every B-operand
// load is a coalesced wave-wide 1KB burst (lane*32B within a contiguous block).

typedef __attribute__((ext_vector_type(16))) __bf16 v16bf;
typedef __attribute__((ext_vector_type(8)))  __bf16 v8bf;
typedef __attribute__((ext_vector_type(8)))  float  v8f;

#define T_STEPS 512
#define BATCH   128
#define IN_DIM  256
#define HID     512
#define ALPHA_F 0.2f
#define LEAK_OVER_A 4.0f   // (1-alpha)/alpha

// ---------------------------------------------------------------------------
// Pack fp32 weights into bf16 WMMA-fragment-major layout.
// For W[n][k] (row-major, pitch = K):
//   dst[(((nt*KC)+kc)*32 + l)*16 + j] = W[nt*16 + (l&15)][kc*32 + (l>>4)*8 + (j&7) + (j>>3)*16]
// so a wave's B fragment for (n-tile nt, k-chunk kc) is one contiguous 1KB
// block, lane l reading 32 contiguous bytes at l*32.
// ---------------------------------------------------------------------------
template <int K, int KC>
__device__ __forceinline__ void pack_frag(const float* __restrict__ W,
                                          __bf16* __restrict__ dst,
                                          int idx, int stride, int total) {
  for (int i = idx; i < total; i += stride) {
    const int j  = i & 15;
    const int l  = (i >> 4) & 31;
    const int kc = (i >> 9) % KC;
    const int nt = (i >> 9) / KC;
    const int n  = nt * 16 + (l & 15);
    const int k  = kc * 32 + (l >> 4) * 8 + (j & 7) + (j >> 3) * 16;
    dst[i] = (__bf16)W[(size_t)n * K + k];
  }
}

__global__ void ctrnn_pack_weights(const float* __restrict__ Win,
                                   const float* __restrict__ Whh,
                                   __bf16* __restrict__ WinP,
                                   __bf16* __restrict__ WhhP) {
  const int idx = blockIdx.x * blockDim.x + threadIdx.x;
  const int stride = gridDim.x * blockDim.x;
  pack_frag<IN_DIM, IN_DIM / 32>(Win, WinP, idx, stride, HID * IN_DIM);
  pack_frag<HID, HID / 32>(Whh, WhhP, idx, stride, HID * HID);
}

// ---------------------------------------------------------------------------
// Input projection: xin[T*B, H] = x[T*B, I] @ W_in^T   (fp32 out, bf16 wmma)
// One workgroup = 16-row slab of x; 8 waves x 4 N-tiles cover all H=512 cols.
// ---------------------------------------------------------------------------
#define XPITCH 264   // 256 + 8 bf16 pad: 528B rows, 16B aligned, bank-rotated

__global__ __launch_bounds__(256) void ctrnn_input_proj(
    const float* __restrict__ x, const __bf16* __restrict__ WinP,
    float* __restrict__ xin) {
  __shared__ __bf16 xs[16][XPITCH];

  const int tid = threadIdx.x;
  const int rr0 = blockIdx.x * 16;          // flattened (t*B+b) row base

  // Stage + convert the 16x256 fp32 x slab into bf16 LDS (coalesced).
  #pragma unroll
  for (int i = 0; i < 16; ++i) {
    const int e = tid + i * 256;
    const int row = e >> 8, col = e & 255;
    xs[row][col] = (__bf16)x[(size_t)(rr0 + row) * IN_DIM + col];
  }
  __syncthreads();

  const int lane  = tid & 31;
  const int wave  = tid >> 5;
  const int lmod  = lane & 15;
  const int lhalf = lane >> 4;
  const int khalf = lhalf * 8;

  #pragma unroll
  for (int tt = 0; tt < 4; ++tt) {
    const int nt = wave * 4 + tt;
    const int n0 = nt * 16;
    v8f acc = {};
    #pragma unroll
    for (int kc = 0; kc < IN_DIM / 32; ++kc) {
      const int kb = kc * 32;
      // A operand (16x32 bf16) from LDS: lane=M, two 16B chunks.
      const __bf16* ap = &xs[lmod][kb + khalf];
      v8bf alo = *reinterpret_cast<const v8bf*>(ap);
      v8bf ahi = *reinterpret_cast<const v8bf*>(ap + 16);
      v16bf a;
      #pragma unroll
      for (int i = 0; i < 8; ++i) { a[i] = alo[i]; a[i + 8] = ahi[i]; }
      // B operand: contiguous fragment, lane*32B -> fully coalesced 1KB/wave.
      v16bf b = *reinterpret_cast<const v16bf*>(
          WinP + (size_t)((nt * (IN_DIM / 32) + kc) * 32 + lane) * 16);
      acc = __builtin_amdgcn_wmma_f32_16x16x32_bf16(
          false, a, false, b, (short)0, acc, false, false);
    }
    // D layout: VGPR r -> M = r + lhalf*8, N = lmod.
    #pragma unroll
    for (int r = 0; r < 8; ++r) {
      const int m = r + lhalf * 8;
      xin[(size_t)(rr0 + m) * HID + n0 + lmod] = acc[r];
    }
  }
}

// ---------------------------------------------------------------------------
// Recurrence: each workgroup owns 16 batch rows for ALL 512 timesteps.
// fp32 h lives in registers (lane-owned per C/D layout); bf16 shadow in LDS
// (double-buffered) is the shared WMMA A-operand. One barrier per step.
// h_new = relu(alpha * ( ((1-a)/a)*h + xin + h @ Whh^T ))
// ---------------------------------------------------------------------------
#define HPITCH 520   // 512 + 8 bf16 pad: 1040B rows, 16B aligned, bank-rotated

__global__ __launch_bounds__(256) void ctrnn_recurrence(
    const float* __restrict__ hidden, const __bf16* __restrict__ WhhP,
    const float* __restrict__ xin, float* __restrict__ out) {
  __shared__ __bf16 hbf[2][16][HPITCH];

  const int tid   = threadIdx.x;
  const int b0    = blockIdx.x * 16;
  const int lane  = tid & 31;
  const int wave  = tid >> 5;
  const int lmod  = lane & 15;
  const int lhalf = lane >> 4;
  const int khalf = lhalf * 8;

  // Persistent fp32 hidden state: 4 tiles x 8 elements per lane.
  v8f hreg[4];
  #pragma unroll
  for (int tt = 0; tt < 4; ++tt) {
    const int n0 = (wave * 4 + tt) * 16;
    #pragma unroll
    for (int r = 0; r < 8; ++r) {
      const int m = r + lhalf * 8;
      const float hv = hidden[(size_t)(b0 + m) * HID + n0 + lmod];
      hreg[tt][r] = hv;
      hbf[0][m][n0 + lmod] = (__bf16)hv;
    }
  }
  __syncthreads();

  int cur = 0;
  for (int t = 0; t < T_STEPS; ++t) {
    const int nxt = cur ^ 1;
    const size_t xrow = (size_t)t * BATCH + b0;

    #pragma unroll
    for (int tt = 0; tt < 4; ++tt) {
      const int nt = wave * 4 + tt;
      const int n0 = nt * 16;

      // Accumulator seeded with leak + input projection (scaled by 1/alpha).
      v8f acc;
      #pragma unroll
      for (int r = 0; r < 8; ++r) {
        const int m = r + lhalf * 8;
        acc[r] = LEAK_OVER_A * hreg[tt][r] +
                 xin[(xrow + m) * HID + n0 + lmod];
      }

      #pragma unroll
      for (int kc = 0; kc < HID / 32; ++kc) {
        const int kb = kc * 32;
        const __bf16* ap = &hbf[cur][lmod][kb + khalf];
        v8bf alo = *reinterpret_cast<const v8bf*>(ap);
        v8bf ahi = *reinterpret_cast<const v8bf*>(ap + 16);
        v16bf a;
        #pragma unroll
        for (int i = 0; i < 8; ++i) { a[i] = alo[i]; a[i + 8] = ahi[i]; }
        // Coalesced fragment-major W_hh load (1KB per wave per wmma).
        v16bf b = *reinterpret_cast<const v16bf*>(
            WhhP + (size_t)((nt * (HID / 32) + kc) * 32 + lane) * 16);
        acc = __builtin_amdgcn_wmma_f32_16x16x32_bf16(
            false, a, false, b, (short)0, acc, false, false);
      }

      // h_new = relu(alpha * acc); update registers, LDS shadow, and output.
      #pragma unroll
      for (int r = 0; r < 8; ++r) {
        const int m = r + lhalf * 8;
        float hn = ALPHA_F * acc[r];
        hn = hn > 0.0f ? hn : 0.0f;
        hreg[tt][r] = hn;
        hbf[nxt][m][n0 + lmod] = (__bf16)hn;
        out[(xrow + m) * HID + n0 + lmod] = hn;
      }
    }

    // Prefetch next step's 16x512 fp32 xin tile (256 threads x 128B = 32KB)
    // so the barrier doesn't serialize the next C-seed loads.
    if (t + 1 < T_STEPS) {
      const float* pf = xin + ((size_t)(t + 1) * BATCH + b0 + (tid >> 4)) * HID
                        + (tid & 15) * 32;
      __builtin_prefetch(pf, 0, 1);
    }

    __syncthreads();   // all waves wrote hbf[nxt]; safe to recycle hbf[cur]
    cur = nxt;
  }

  // h_final appended after output tensor.
  float* hfin = out + (size_t)T_STEPS * BATCH * HID;
  #pragma unroll
  for (int tt = 0; tt < 4; ++tt) {
    const int n0 = (wave * 4 + tt) * 16;
    #pragma unroll
    for (int r = 0; r < 8; ++r) {
      const int m = r + lhalf * 8;
      hfin[(size_t)(b0 + m) * HID + n0 + lmod] = hreg[tt][r];
    }
  }
}

// ---------------------------------------------------------------------------
extern "C" void kernel_launch(void* const* d_in, const int* in_sizes, int n_in,
                              void* d_out, int out_size, void* d_ws, size_t ws_size,
                              hipStream_t stream) {
  const float* x      = (const float*)d_in[0];  // [T,B,I]
  const float* hidden = (const float*)d_in[1];  // [B,H]
  const float* W_in   = (const float*)d_in[2];  // [H,I]
  const float* W_hh   = (const float*)d_in[3];  // [H,H]
  float* out = (float*)d_out;                   // [T,B,H] ++ [B,H]

  // Workspace layout (offsets 256B-aligned):
  //   [0)            WhhP : 512*512*2 = 524288 B (fragment-major bf16)
  //   [524288)       WinP : 512*256*2 = 262144 B (fragment-major bf16)
  //   [786432)       xin  : 65536*512*4 fp32
  char* ws = (char*)d_ws;
  __bf16* WhhP = (__bf16*)(ws);
  __bf16* WinP = (__bf16*)(ws + 524288);
  float*  xin  = (float*)(ws + 786432);

  ctrnn_pack_weights<<<512, 256, 0, stream>>>(W_in, W_hh, WinP, WhhP);
  ctrnn_input_proj<<<(T_STEPS * BATCH) / 16, 256, 0, stream>>>(x, WinP, xin);
  ctrnn_recurrence<<<BATCH / 16, 256, 0, stream>>>(hidden, WhhP, xin, out);
}
